// LinearAttention_Dec_70583492542798
// MI455X (gfx1250) — matrix operand order repaired and checked by reading, not compiled, not run on hardware
//
#include <hip/hip_runtime.h>

// ---------------------------------------------------------------------------
// LinearAttention (RoPE linear attention + LePE) for MI455X / gfx1250
//   K0 wcvt     : Wq,Wk f32 -> f16
//   K1 proj<K>  : k = elu(y@Wk^T+bk)+1 ; rope -> kro(f16) ; col-sum partials
//   K2 ksum_red : deterministic reduce of k column sums (for kmean)
//   K3 proj<Q>  : q = elu(x@Wq^T+bq)+1 ; z = 1/(q.kmean+1e-6) ; rope -> qro(f16)
//   K4 kv       : kv partials = k_rope^T @ v per (b,h,nseg); k tile staged via
//                 TDM tensor_load_to_lds (s_wait_tensorcnt) when available
//   K5 kv_red   : sum partials, *1/N, store kv TRANSPOSED f16 [bh][e][d]
//   K6 out      : out = (q_rope @ kv) * z + depthwise-conv3(y) + bias
// proj tiling: 32 rows x 512 cols per block so x/y stream from HBM exactly once.
// ---------------------------------------------------------------------------

typedef __attribute__((ext_vector_type(16))) _Float16 v16h;
typedef __attribute__((ext_vector_type(8)))  _Float16 v8h;
typedef __attribute__((ext_vector_type(8)))  float    v8f;
typedef unsigned int  u32x4 __attribute__((ext_vector_type(4)));
typedef int           i32x8 __attribute__((ext_vector_type(8)));
typedef int           i32x4 __attribute__((ext_vector_type(4)));

#if defined(__has_builtin)
#if __has_builtin(__builtin_amdgcn_tensor_load_to_lds) && \
    __has_builtin(__builtin_amdgcn_s_wait_tensorcnt)
#define HAS_TDM 1
#endif
#endif
#ifndef HAS_TDM
#define HAS_TDM 0
#endif

#define DEV __device__ __forceinline__

constexpr int Bc = 8;            // batch
constexpr int Nn = 4096;         // sequence
constexpr int Cc = 512;          // channels
constexpr int Hh = 8;            // heads
constexpr int Dh = 64;           // head dim
constexpr int BN = Bc * Nn;      // 32768 rows
constexpr int RT = 32;           // proj row tile per block
constexpr float ROPE_LTH = 0.03597789207808088f;  // ln(10000)/256

DEV int laneid() { return threadIdx.x & 31; }

DEV v16h make16(v8h lo, v8h hi) {
  v16h r;
#pragma unroll
  for (int i = 0; i < 8; ++i) { r[i] = lo[i]; r[i + 8] = hi[i]; }
  return r;
}

// A-fragment (16x32 f16) from f32 row-major (ISA 16-bit A layout:
// lanes 0-15 -> K {0..7,16..23}, lanes 16-31 -> K {8..15,24..31})
DEV v16h loadA_cvt(const float* __restrict__ base, size_t row, int ld, int k0) {
  const int h0 = (laneid() < 16) ? 0 : 8;
  const float* p = base + row * (size_t)ld + k0 + h0;
  v16h r;
#pragma unroll
  for (int i = 0; i < 8; ++i) { r[i] = (_Float16)p[i]; r[i + 8] = (_Float16)p[16 + i]; }
  return r;
}

// A-fragment from f16 row-major
DEV v16h loadA_h(const _Float16* __restrict__ base, size_t row, int ld, int k0) {
  const int h0 = (laneid() < 16) ? 0 : 8;
  const _Float16* p = base + row * (size_t)ld + k0 + h0;
  v8h lo = *(const v8h*)p;
  v8h hi = *(const v8h*)(p + 16);
  return make16(lo, hi);
}

// B-fragment (32x16 f16): colptr = start of this lane's column (K-contiguous);
// lanes 0-15 hold K {0..7,16..23}, lanes 16-31 hold K {8..15,24..31}.
DEV v16h loadB_h(const _Float16* __restrict__ colptr, int k0) {
  const _Float16* p = colptr + k0 + ((laneid() < 16) ? 0 : 8);
  v8h lo = *(const v8h*)p;
  v8h hi = *(const v8h*)(p + 16);
  return make16(lo, hi);
}

DEV v8f wmma_f16(v16h a, v16h b, v8f c) {
  return __builtin_amdgcn_wmma_f32_16x16x32_f16(false, a, false, b, (short)0, c,
                                                false, false);
}

// ---------------------------------------------------------------------------
// K0: weight f32 -> f16 conversion
__global__ __launch_bounds__(256)
void wcvt_kernel(const float* __restrict__ Wq, const float* __restrict__ Wk,
                 _Float16* __restrict__ wqh, _Float16* __restrict__ wkh) {
  int idx = blockIdx.x * 256 + threadIdx.x;
  if (idx < Cc * Cc) {
    wqh[idx] = (_Float16)Wq[idx];
    wkh[idx] = (_Float16)Wk[idx];
  }
}

// ---------------------------------------------------------------------------
// K1/K3: projection GEMM. Block = 32 rows x 512 cols, 8 waves:
// wave w -> rows (w&1)*16, cols (w>>1)*128. x/y read exactly once from HBM.
template <bool IS_Q>
__global__ __launch_bounds__(256)
void proj_kernel(const float* __restrict__ X, const _Float16* __restrict__ Wh,
                 const float* __restrict__ bias, _Float16* __restrict__ ro,
                 float* __restrict__ pk, const float* __restrict__ ksum,
                 float* __restrict__ z) {
  __shared__ float lds_ps[2][Cc];
  const int lane = laneid();
  const int wave = threadIdx.x >> 5;
  const int rw = wave & 1, cw = wave >> 1;
  const int rb = blockIdx.x;
  const int row0 = rb * RT + rw * 16;
  const int col0 = cw * 128;
  const size_t rowA = (size_t)row0 + (lane & 15);
  const int b = (rb * RT) >> 12;  // all 32 rows share one batch

  v8f acc[8] = {};
#pragma unroll 2
  for (int k0 = 0; k0 < Cc; k0 += 32) {
    if (k0 + 32 < Cc)
      __builtin_prefetch(X + rowA * Cc + k0 + 32, 0, 1);  // global_prefetch_b8
    v16h a = loadA_cvt(X, rowA, Cc, k0);
#pragma unroll
    for (int ef = 0; ef < 8; ++ef) {
      int col = col0 + ef * 16 + (lane & 15);
      v16h bf = loadB_h(Wh + (size_t)col * Cc, k0);
      acc[ef] = wmma_f16(a, bf, acc[ef]);
    }
  }

  // bias + elu + 1
#pragma unroll
  for (int ef = 0; ef < 8; ++ef) {
    int col = col0 + ef * 16 + (lane & 15);
    float bv = bias[col];
#pragma unroll
    for (int vr = 0; vr < 8; ++vr) {
      float v = acc[ef][vr] + bv;
      acc[ef][vr] = (v > 0.0f) ? (v + 1.0f) : __expf(v);
    }
  }

  if constexpr (!IS_Q) {
    // deterministic per-block column sums (pre-rope k, for kmean)
#pragma unroll
    for (int ef = 0; ef < 8; ++ef) {
      float s = 0.f;
#pragma unroll
      for (int vr = 0; vr < 8; ++vr) s += acc[ef][vr];
      s += __shfl_xor(s, 16, 32);
      if (lane < 16) lds_ps[rw][col0 + ef * 16 + lane] = s;
    }
    __syncthreads();
    for (int c = threadIdx.x; c < Cc; c += 256)
      pk[(size_t)rb * Cc + c] = lds_ps[0][c] + lds_ps[1][c];
  } else {
    // z = 1/(q . kmean + 1e-6), two heads per wave
    float kmv[8];
#pragma unroll
    for (int ef = 0; ef < 8; ++ef) {
      int col = col0 + ef * 16 + (lane & 15);
      kmv[ef] = ksum[b * Cc + col] * (1.0f / (float)Nn);
    }
#pragma unroll
    for (int hh = 0; hh < 2; ++hh) {
      int h = cw * 2 + hh;
#pragma unroll
      for (int vr = 0; vr < 8; ++vr) {
        float dot = 0.f;
#pragma unroll
        for (int e4 = 0; e4 < 4; ++e4) dot += acc[hh * 4 + e4][vr] * kmv[hh * 4 + e4];
        dot += __shfl_xor(dot, 1, 32);
        dot += __shfl_xor(dot, 2, 32);
        dot += __shfl_xor(dot, 4, 32);
        dot += __shfl_xor(dot, 8, 32);
        if ((lane & 15) == 0) {
          int rowg = row0 + vr + ((lane < 16) ? 0 : 8);
          int n = rowg & (Nn - 1);
          z[((size_t)(b * Hh + h)) * Nn + n] = 1.0f / (dot + 1e-6f);
        }
      }
    }
  }

  // RoPE + f16 store (lane pairs hold adjacent channels -> shfl_xor(1))
#pragma unroll
  for (int ef = 0; ef < 8; ++ef) {
    int col = col0 + ef * 16 + (lane & 15);
    float theta = __expf(-(float)(col >> 1) * ROPE_LTH);
    bool is_r = (col & 1) == 0;
#pragma unroll
    for (int vr = 0; vr < 8; ++vr) {
      int rowg = row0 + vr + ((lane < 16) ? 0 : 8);
      int n = rowg & (Nn - 1);
      float sn, cs;
      __sincosf((float)n * theta, &sn, &cs);
      float av = acc[ef][vr];
      float pv = __shfl_xor(av, 1, 32);
      float o = is_r ? (cs * av - sn * pv) : (sn * pv + cs * av);
      ro[(size_t)rowg * Cc + col] = (_Float16)o;
    }
  }
}

// ---------------------------------------------------------------------------
// K2: reduce k column-sum partials (128 row-blocks per batch)
__global__ __launch_bounds__(256)
void ksum_kernel(const float* __restrict__ pk, float* __restrict__ ksum) {
  int idx = blockIdx.x * 256 + threadIdx.x;  // Bc*Cc = 4096
  int b = idx >> 9, c = idx & 511;
  float s = 0.f;
#pragma unroll 8
  for (int i = 0; i < BN / RT / Bc; ++i) s += pk[(size_t)(b * (BN / RT / Bc) + i) * Cc + c];
  ksum[idx] = s;
}

// ---------------------------------------------------------------------------
// K4: kv partials per (b,h,seg): kv[d][e] = sum_n k_rope[n,d]*v[n,e]
// k segment (512 x 64 f16 = 64KB) staged in LDS via the Tensor Data Mover.
__global__ __launch_bounds__(256)
void kv_kernel(const _Float16* __restrict__ kro, const float* __restrict__ Y,
               float* __restrict__ kvpart) {
  __shared__ __attribute__((aligned(16))) _Float16 kS[512][Dh];  // [n][d], 64 KB
  __shared__ __attribute__((aligned(16))) _Float16 vT[Dh][32];   // [e][n]
  const int lane = laneid();
  const int wave = threadIdx.x >> 5;
  const int bh = blockIdx.x, b = bh >> 3, h = bh & 7;
  const int seg = blockIdx.y;
  const size_t gbase = ((size_t)(b * Nn + seg * 512)) * Cc + h * Dh;

#if HAS_TDM
  if (wave == 0) {
    unsigned long long ga = (unsigned long long)(const void*)(kro + gbase);
    unsigned int lds_base = (unsigned int)(unsigned long long)(const void*)&kS[0][0];
    u32x4 g0;
    g0[0] = 1u;                                    // count=1, user descriptor
    g0[1] = lds_base;                              // lds_addr
    g0[2] = (unsigned int)(ga & 0xffffffffu);      // global_addr[31:0]
    g0[3] = (unsigned int)((ga >> 32) & 0x01ffffffu) | (2u << 30);  // addr[56:32], type=2
    i32x8 g1;
    g1[0] = (int)(1u << 16);                       // wg_mask=0, data_size=1 (2B)
    g1[1] = (int)((Cc & 0xffffu) << 16);           // tensor_dim0 lo16 at bit 48
    g1[2] = (int)((Cc >> 16) | ((BN & 0xffffu) << 16));   // dim0 hi | dim1 lo
    g1[3] = (int)((BN >> 16) | (Dh << 16));        // dim1 hi | tile_dim0=64
    g1[4] = (int)(512);                            // tile_dim1=512, tile_dim2=0
    g1[5] = (int)(Cc);                             // tensor_dim0_stride lo32
    g1[6] = 0;                                     // stride0 hi | stride1 lo
    g1[7] = 0;
    i32x4 zg4; zg4[0] = 0; zg4[1] = 0; zg4[2] = 0; zg4[3] = 0;
    i32x8 zg8; zg8[0] = 0; zg8[1] = 0; zg8[2] = 0; zg8[3] = 0;
    zg8[4] = 0; zg8[5] = 0; zg8[6] = 0; zg8[7] = 0;
    __builtin_amdgcn_tensor_load_to_lds(g0, g1, zg4, zg4, zg8, 0);
    __builtin_amdgcn_s_wait_tensorcnt((unsigned short)0);
  }
  __syncthreads();
#else
  for (int i = threadIdx.x; i < 512 * Dh / 8; i += 256) {
    int r = i >> 3, c0 = (i & 7) * 8;
    *(v8h*)&kS[r][c0] = *(const v8h*)(kro + gbase + (size_t)r * Cc + c0);
  }
  __syncthreads();
#endif

  v8f acc[2] = {};
  const int p0 = wave * 2;
  for (int s = 0; s < 16; ++s) {
    {  // stage v (f32 -> f16, transposed for B-fragments)
      int r = threadIdx.x >> 3;          // 32 n-rows
      int c0 = (threadIdx.x & 7) * 8;    // 8 channels each
      const float* vs = Y + gbase + (size_t)(s * 32 + r) * Cc + c0;
#pragma unroll
      for (int i = 0; i < 8; ++i) vT[c0 + i][r] = (_Float16)vs[i];
    }
    __syncthreads();
#pragma unroll
    for (int pp = 0; pp < 2; ++pp) {
      int p = p0 + pp, df = p >> 2, ef = p & 3;
      int d = df * 16 + (lane & 15);
      int h0 = (lane < 16) ? 0 : 8;
      v16h a;
#pragma unroll
      for (int i = 0; i < 8; ++i) {       // strided reads from TDM-staged tile
        a[i]     = kS[s * 32 + h0 + i][d];
        a[i + 8] = kS[s * 32 + 16 + h0 + i][d];
      }
      int e = ef * 16 + (lane & 15);
      int bo = (lane < 16) ? 0 : 8;
      v8h blo = *(const v8h*)&vT[e][bo];
      v8h bhi = *(const v8h*)&vT[e][bo + 16];
      acc[pp] = wmma_f16(a, make16(blo, bhi), acc[pp]);
    }
    __syncthreads();
  }
#pragma unroll
  for (int pp = 0; pp < 2; ++pp) {
    int p = p0 + pp, df = p >> 2, ef = p & 3;
    int e = ef * 16 + (lane & 15);
#pragma unroll
    for (int vr = 0; vr < 8; ++vr) {
      int d = df * 16 + vr + ((lane < 16) ? 0 : 8);
      kvpart[(((size_t)seg * 64 + bh) * Dh + d) * Dh + e] = acc[pp][vr];
    }
  }
}

// ---------------------------------------------------------------------------
// K5: reduce kv partials, scale by 1/N, store transposed f16 [bh][e][d]
__global__ __launch_bounds__(256)
void kvred_kernel(const float* __restrict__ kvpart, _Float16* __restrict__ kvT) {
  int bh = blockIdx.x;
  for (int i = threadIdx.x; i < Dh * Dh; i += 256) {
    int d = i >> 6, e = i & 63;
    float s = 0.f;
#pragma unroll
    for (int g = 0; g < 8; ++g)
      s += kvpart[(((size_t)g * 64 + bh) * Dh + d) * Dh + e];
    kvT[((size_t)bh * Dh + e) * Dh + d] = (_Float16)(s * (1.0f / (float)Nn));
  }
}

// ---------------------------------------------------------------------------
// K6: out = (q_rope @ kv) * z + depthwise conv3(y) + bias
__global__ __launch_bounds__(256)
void out_kernel(const _Float16* __restrict__ qro, const _Float16* __restrict__ kvT,
                const float* __restrict__ z, const float* __restrict__ Y,
                const float* __restrict__ w3, const float* __restrict__ lb,
                float* __restrict__ out) {
  const int lane = laneid();
  const int wave = threadIdx.x >> 5;
  const int rb = blockIdx.x, ct = blockIdx.y;
  const int row0 = rb * 128 + wave * 16;
  const int b = (rb * 128) >> 12;
  const size_t rowA = (size_t)row0 + (lane & 15);
  v8f acc[8] = {};
#pragma unroll
  for (int k0 = 0; k0 < Dh; k0 += 32) {
    v16h a0 = loadA_h(qro, rowA, Cc, (ct * 2 + 0) * Dh + k0);
    v16h a1 = loadA_h(qro, rowA, Cc, (ct * 2 + 1) * Dh + k0);
#pragma unroll
    for (int ef = 0; ef < 8; ++ef) {
      int hh = ef >> 2;
      int el = (ef & 3) * 16 + (lane & 15);
      int bhidx = b * Hh + ct * 2 + hh;
      v16h bf = loadB_h(kvT + ((size_t)bhidx * Dh + el) * Dh, k0);
      acc[ef] = wmma_f16(hh ? a1 : a0, bf, acc[ef]);
    }
  }
#pragma unroll
  for (int ef = 0; ef < 8; ++ef) {
    int hh = ef >> 2, h = ct * 2 + hh;
    int c = ct * 128 + ef * 16 + (lane & 15);
    float w0 = w3[c * 3 + 0], w1 = w3[c * 3 + 1], w2 = w3[c * 3 + 2];
    float bb = lb[c];
#pragma unroll
    for (int vr = 0; vr < 8; ++vr) {
      int rowg = row0 + vr + ((lane < 16) ? 0 : 8);
      int n = rowg & (Nn - 1);
      float zv = z[((size_t)(b * Hh + h)) * Nn + n];
      float lep = w1 * Y[(size_t)rowg * Cc + c] + bb;
      if (n > 0) lep += w0 * Y[(size_t)(rowg - 1) * Cc + c];
      if (n < Nn - 1) lep += w2 * Y[(size_t)(rowg + 1) * Cc + c];
      out[(size_t)rowg * Cc + c] = acc[ef][vr] * zv + lep;
    }
  }
}

// ---------------------------------------------------------------------------
extern "C" void kernel_launch(void* const* d_in, const int* in_sizes, int n_in,
                              void* d_out, int out_size, void* d_ws, size_t ws_size,
                              hipStream_t stream) {
  (void)in_sizes; (void)n_in; (void)out_size; (void)ws_size;
  const float* x  = (const float*)d_in[0];
  const float* y  = (const float*)d_in[1];
  const float* Wq = (const float*)d_in[2];
  const float* bq = (const float*)d_in[3];
  const float* Wk = (const float*)d_in[4];
  const float* bk = (const float*)d_in[5];
  const float* lw = (const float*)d_in[6];
  const float* lb = (const float*)d_in[7];
  float* out = (float*)d_out;

  char* ws = (char*)d_ws;
  constexpr size_t OFF_WQH = 0;                              // 512 KB
  constexpr size_t OFF_WKH = OFF_WQH + (size_t)Cc * Cc * 2;  // 512 KB
  constexpr size_t OFF_QRO = OFF_WKH + (size_t)Cc * Cc * 2;  // 32 MB
  constexpr size_t OFF_KRO = OFF_QRO + (size_t)BN * Cc * 2;  // 32 MB
  constexpr size_t OFF_PK  = OFF_KRO + (size_t)BN * Cc * 2;  // 2 MB
  constexpr size_t OFF_KS  = OFF_PK + (size_t)(BN / RT) * Cc * 4;   // 16 KB
  constexpr size_t OFF_Z   = OFF_KS + (size_t)Bc * Cc * 4;          // 1 MB
  constexpr size_t OFF_KVP = OFF_Z + (size_t)Bc * Hh * Nn * 4;      // 8 MB
  constexpr size_t OFF_KVT = OFF_KVP + (size_t)8 * Bc * Hh * Dh * Dh * 4;  // 512 KB

  _Float16* wqh = (_Float16*)(ws + OFF_WQH);
  _Float16* wkh = (_Float16*)(ws + OFF_WKH);
  _Float16* qro = (_Float16*)(ws + OFF_QRO);
  _Float16* kro = (_Float16*)(ws + OFF_KRO);
  float* pk     = (float*)(ws + OFF_PK);
  float* ksum   = (float*)(ws + OFF_KS);
  float* zbuf   = (float*)(ws + OFF_Z);
  float* kvpart = (float*)(ws + OFF_KVP);
  _Float16* kvT = (_Float16*)(ws + OFF_KVT);

  dim3 blk(256);
  wcvt_kernel<<<dim3((Cc * Cc + 255) / 256), blk, 0, stream>>>(Wq, Wk, wqh, wkh);
  proj_kernel<false><<<dim3(BN / RT), blk, 0, stream>>>(
      y, wkh, bk, kro, pk, nullptr, nullptr);
  ksum_kernel<<<dim3((Bc * Cc) / 256), blk, 0, stream>>>(pk, ksum);
  proj_kernel<true><<<dim3(BN / RT), blk, 0, stream>>>(
      x, wqh, bq, qro, nullptr, ksum, zbuf);
  kv_kernel<<<dim3(Bc * Hh, 8), blk, 0, stream>>>(kro, y, kvpart);
  kvred_kernel<<<dim3(Bc * Hh), blk, 0, stream>>>(kvpart, kvT);
  out_kernel<<<dim3(BN / 128, Cc / 128), blk, 0, stream>>>(
      qro, kvT, zbuf, y, lw, lb, out);
}
